// DynaAttentionBlock_64115271794954
// MI455X (gfx1250) — matrix operand order repaired and checked by reading, not compiled, not run on hardware
//
#include <hip/hip_runtime.h>

// Problem constants (match reference)
#define B_    4
#define FIN_  256
#define N_    2048
#define LAT_  128
#define H_    8
#define FH_   64
#define HF_   512
#define TEMP_ 0.35355339059327373f  // (FH/H)^-0.5 = 8^-0.5

typedef __attribute__((ext_vector_type(2)))  float  v2f;
typedef __attribute__((ext_vector_type(8)))  float  v8f;
typedef __attribute__((ext_vector_type(16))) __bf16 v16bf;
typedef __attribute__((ext_vector_type(8)))  __bf16 v8bf;

static __device__ __forceinline__ __bf16 to_bf16(float f) {
  unsigned u = __builtin_bit_cast(unsigned, f);
  u += 0x7FFFu + ((u >> 16) & 1u);              // round to nearest even
  unsigned short s = (unsigned short)(u >> 16);
  return __builtin_bit_cast(__bf16, s);
}

// XOR-butterfly via ds_swizzle (compile-time pattern, no index-setup VALU).
// group-of-32 mode: and_mask=0x1f, or_mask=0, xor_mask=XORM -> lane ^ XORM.
template <int XORM>
static __device__ __forceinline__ float swz_xor(float v) {
  int i = __builtin_bit_cast(int, v);
  i = __builtin_amdgcn_ds_swizzle(i, 0x1f | (XORM << 10));
  return __builtin_bit_cast(float, i);
}
static __device__ __forceinline__ float red_max16(float v) {
  v = fmaxf(v, swz_xor<1>(v));
  v = fmaxf(v, swz_xor<2>(v));
  v = fmaxf(v, swz_xor<4>(v));
  v = fmaxf(v, swz_xor<8>(v));
  return v;  // reduced across each 16-lane half
}
static __device__ __forceinline__ float red_sum16(float v) {
  v += swz_xor<1>(v);
  v += swz_xor<2>(v);
  v += swz_xor<4>(v);
  v += swz_xor<8>(v);
  return v;
}

// 32-bit LDS byte offset of a generic pointer that points into LDS.
static __device__ __forceinline__ unsigned lds_off(const void* p) {
  return (unsigned)(size_t)(const __attribute__((address_space(3))) void*)p;
}

// Async copy 16B/lane from global to LDS (ASYNCcnt-tracked, CDNA5 path).
#define ASYNC_LD_B128(ldsoff, gaddr)                                 \
  asm volatile("global_load_async_to_lds_b128 %0, %1, off"           \
               :: "v"(ldsoff), "v"(gaddr) : "memory")

#define WAIT_ASYNCCNT0() asm volatile("s_wait_asynccnt 0x0" ::: "memory")

// ---------------------------------------------------------------------------
// Kernel 1: hypernet weight generation  out[b][j] = sum_l lat[b][l]*Wg[l][j]
// Memory-bound stream of ~268MB of Wg => this is the runtime floor
// (268MB / 23.3TB/s ~= 11.5us). Coalesced j-major reads, lat in LDS.
// ---------------------------------------------------------------------------
__global__ void gen_w_kernel(const float* __restrict__ lat,
                             const float* __restrict__ Wg,
                             float* __restrict__ out, int J) {
  __shared__ float ls[LAT_];
  const int b = blockIdx.y;
  if (threadIdx.x < LAT_) ls[threadIdx.x] = lat[b * LAT_ + threadIdx.x];
  __syncthreads();
  const int j = blockIdx.x * blockDim.x + threadIdx.x;
  if (j >= J) return;
  const float* wp = Wg + j;
  float acc = 0.f;
#pragma unroll 8
  for (int l = 0; l < LAT_; ++l) acc = __builtin_fmaf(ls[l], wp[(size_t)l * J], acc);
  out[(size_t)b * J + j] = acc;
}

// ---------------------------------------------------------------------------
// Kernel 2: dynamic 1x1 conv q/k/v:  C[512,2048] = W[512,256] @ X[256,2048]+b
// fp32 WMMA 16x16x4; W (2MB) and X (8MB) are L2-resident (192MB L2).
// MODE 0: q -> bf16 [b,h,n,64] scaled by TEMP
// MODE 1: k -> bf16 [b,h,n,64]
// MODE 2: v -> bf16 [b,512,n] (feature-major)
// ---------------------------------------------------------------------------
template <int MODE>
__global__ void proj_qkv_kernel(const float* __restrict__ W,
                                const float* __restrict__ bias,
                                const float* __restrict__ X,
                                unsigned short* __restrict__ out_raw) {
  __bf16* __restrict__ out = reinterpret_cast<__bf16*>(out_raw);
  const int b    = blockIdx.z;
  const int m0   = blockIdx.y * 16;
  const int wave = threadIdx.x >> 5;
  const int n0   = blockIdx.x * 64 + wave * 16;
  const int lane = threadIdx.x & 31;
  const int half = lane >> 4, l16 = lane & 15;

  // A: 16x4 f32 frag: lanes<16 hold K={k,k+1}, lanes>=16 hold K={k+2,k+3}
  const float* Wb = W + (size_t)b * HF_ * FIN_ + (size_t)(m0 + l16) * FIN_ + 2 * half;
  // B: 4x16 f32 frag: lane col = n0+l16; VGPR0: K=k (half0)/k+2 (half1), VGPR1: +1
  const float* Xb = X + (size_t)b * FIN_ * N_ + (size_t)(2 * half) * N_ + n0 + l16;

  v8f c = {};
#pragma unroll 4
  for (int k = 0; k < FIN_; k += 4) {
    v2f a, bb;
    a.x  = Wb[k];
    a.y  = Wb[k + 1];
    bb.x = Xb[(size_t)k * N_];
    bb.y = Xb[(size_t)(k + 1) * N_];
    c = __builtin_amdgcn_wmma_f32_16x16x4_f32(false, a, false, bb, (short)0, c,
                                              false, false);
  }
#pragma unroll
  for (int r = 0; r < 8; ++r) {
    const int f = m0 + r + 8 * half;       // C layout: VGPR r -> M=r (+8 for hi half)
    float v = c[r] + bias[b * HF_ + f];
    if (MODE == 0) v *= TEMP_;             // fold softmax temperature into q
    if (MODE <= 1) {
      const int h = f >> 6, fl = f & 63;
      out[(((size_t)b * H_ + h) * N_ + (n0 + l16)) * FH_ + fl] = to_bf16(v);
    } else {
      out[((size_t)b * HF_ + f) * N_ + n0 + l16] = to_bf16(v);
    }
  }
}

// ---------------------------------------------------------------------------
// Kernel 3: flash attention per (b,h). Block = 4 waves x 16 queries = 64 q.
// K/V chunks (32 keys) are staged into LDS ONCE per block via
// GLOBAL_LOAD_ASYNC_TO_LDS_B128 (double-buffered, overlapped with compute),
// instead of 4x-redundant per-wave global loads. bf16 WMMA 16x16x32 with
// f32 accumulators, online softmax (ds_swizzle butterflies), P flip via LDS,
// output transpose via LDS for coalesced feature-major stores.
// ---------------------------------------------------------------------------
__global__ void attn_kernel(const unsigned short* __restrict__ qt_raw,
                            const unsigned short* __restrict__ kt_raw,
                            const unsigned short* __restrict__ vf_raw,
                            float* __restrict__ o_attn) {
  const __bf16* q_t  = reinterpret_cast<const __bf16*>(qt_raw);
  const __bf16* v_fm = reinterpret_cast<const __bf16*>(vf_raw);

  __shared__ __bf16 kbuf[2][32][64];      // staged K chunk: [key][feature]
  __shared__ __bf16 vbuf[2][64][32];      // staged V chunk: [feature][key]
  __shared__ __bf16 p_lds[4][16][32];     // per-wave P tile (16q x 32k)
  __shared__ float  ot_lds[4][FH_][17];   // per-wave output transpose (pad 17)

  const int b    = blockIdx.z, h = blockIdx.y;
  const int tid  = threadIdx.x;
  const int wave = tid >> 5;
  const int q0   = blockIdx.x * 64 + wave * 16;
  const int lane = tid & 31;
  const int half = lane >> 4, l16 = lane & 15;

  const __bf16* qb = q_t + ((size_t)b * H_ + h) * N_ * FH_;
  const char*   kg = (const char*)(kt_raw + ((size_t)b * H_ + h) * N_ * FH_);
  const char*   vg = (const char*)(v_fm + ((size_t)b * HF_ + h * FH_) * N_);

  // ---- async stage of one 32-key chunk (8KB: 4KB K + 4KB V) ----
  auto stage = [&](int key0, int buf) {
    // K rows key0..key0+31 are contiguous: 32*64*2 = 4096 bytes
    const unsigned kb0 = lds_off(&kbuf[buf][0][0]);
    const char* kp = kg + (size_t)key0 * (FH_ * 2);
    ASYNC_LD_B128(kb0 + (unsigned)(tid * 16), kp + tid * 16);
    ASYNC_LD_B128(kb0 + (unsigned)(2048 + tid * 16), kp + 2048 + tid * 16);
    // V: 64 feature-rows of 64B at column key0 (row stride N_*2 bytes)
    const unsigned vb0 = lds_off(&vbuf[buf][0][0]);
#pragma unroll
    for (int j = 0; j < 2; ++j) {
      const int idx = j * 128 + tid;       // 256 x 16B chunks
      const int f = idx >> 2, c16 = idx & 3;
      ASYNC_LD_B128(vb0 + (unsigned)(f * 64 + c16 * 16),
                    vg + (size_t)f * (N_ * 2) + (size_t)key0 * 2 + c16 * 16);
    }
  };

  // Q A-frags (16x32 bf16), kept in registers for the whole key loop.
  // A layout: lanes<16: elems0-7 = K0-7, elems8-15 = K16-23; lanes>=16: 8-15,24-31
  v16bf aq0, aq1;
  {
    const __bf16* qr = qb + (size_t)(q0 + l16) * FH_;
    const int e0 = half * 8;
    v8bf t0 = *(const v8bf*)(qr + e0);
    v8bf t1 = *(const v8bf*)(qr + 16 + e0);
    v8bf t2 = *(const v8bf*)(qr + 32 + e0);
    v8bf t3 = *(const v8bf*)(qr + 48 + e0);
#pragma unroll
    for (int e = 0; e < 8; ++e) {
      aq0[e] = t0[e]; aq0[8 + e] = t1[e];
      aq1[e] = t2[e]; aq1[8 + e] = t3[e];
    }
  }

  float m[8], lsum[8];
  v8f oacc[4];
#pragma unroll
  for (int r = 0; r < 8; ++r) { m[r] = -1e30f; lsum[r] = 0.f; }
#pragma unroll
  for (int g = 0; g < 4; ++g) oacc[g] = (v8f){};

  stage(0, 0);  // prologue: chunk 0 in flight

  for (int ci = 0; ci < N_ / 32; ++ci) {
    const int buf = ci & 1;
    WAIT_ASYNCCNT0();
    __syncthreads();                      // chunk ci visible to all waves
    if (ci + 1 < N_ / 32) stage((ci + 1) * 32, buf ^ 1);  // overlap next copy

    // ---- S = Q^T K for 16q x 32keys (two 16x16 C frags), K from LDS ----
    v8f s_lo = {}, s_hi = {};
    const __bf16* kr_lo = &kbuf[buf][l16][half * 8];
    const __bf16* kr_hi = &kbuf[buf][l16 + 16][half * 8];
    v16bf bk;
#pragma unroll
    for (int kk = 0; kk < 2; ++kk) {       // feature halves 0-31, 32-63
      const int fo = kk * 32;
      v8bf u0 = *(const v8bf*)(kr_lo + fo);
      v8bf u1 = *(const v8bf*)(kr_lo + fo + 16);
#pragma unroll
      for (int e = 0; e < 8; ++e) { bk[e] = u0[e]; bk[8 + e] = u1[e]; }
      s_lo = __builtin_amdgcn_wmma_f32_16x16x32_bf16(false, kk ? aq1 : aq0, false,
                                                     bk, (short)0, s_lo, false, false);
      u0 = *(const v8bf*)(kr_hi + fo);
      u1 = *(const v8bf*)(kr_hi + fo + 16);
#pragma unroll
      for (int e = 0; e < 8; ++e) { bk[e] = u0[e]; bk[8 + e] = u1[e]; }
      s_hi = __builtin_amdgcn_wmma_f32_16x16x32_bf16(false, kk ? aq1 : aq0, false,
                                                     bk, (short)0, s_hi, false, false);
    }

    // ---- online softmax update (rows = queries; 16-lane ds_swizzle trees) ----
#pragma unroll
    for (int r = 0; r < 8; ++r) {
      const float s0 = s_lo[r], s1 = s_hi[r];
      const float mr = red_max16(fmaxf(s0, s1));
      const float mn = fmaxf(m[r], mr);
      const float sc = __expf(m[r] - mn);
      const float p0 = __expf(s0 - mn);
      const float p1 = __expf(s1 - mn);
      const float rs = red_sum16(p0 + p1);
      lsum[r] = lsum[r] * sc + rs;
      m[r] = mn;
#pragma unroll
      for (int g = 0; g < 4; ++g) oacc[g][r] *= sc;
      p_lds[wave][r + 8 * half][l16]      = to_bf16(p0);
      p_lds[wave][r + 8 * half][l16 + 16] = to_bf16(p1);
    }

    // ---- A frag of P (C-layout -> A-layout via LDS, same-wave in-order) ----
    v16bf ap;
    {
      const __bf16* pr = &p_lds[wave][l16][0];
      v8bf c0 = *(const v8bf*)(pr + half * 8);
      v8bf c1 = *(const v8bf*)(pr + 16 + half * 8);
#pragma unroll
      for (int e = 0; e < 8; ++e) { ap[e] = c0[e]; ap[8 + e] = c1[e]; }
    }

    // ---- O += P @ V (4 feature tiles of 16), V from LDS ----
#pragma unroll
    for (int g = 0; g < 4; ++g) {
      const __bf16* vr = &vbuf[buf][g * 16 + l16][half * 8];
      v8bf u0 = *(const v8bf*)(vr);
      v8bf u1 = *(const v8bf*)(vr + 16);
      v16bf bv;
#pragma unroll
      for (int e = 0; e < 8; ++e) { bv[e] = u0[e]; bv[8 + e] = u1[e]; }
      oacc[g] = __builtin_amdgcn_wmma_f32_16x16x32_bf16(false, ap, false, bv,
                                                        (short)0, oacc[g], false, false);
    }
    __syncthreads();                      // all reads of buf done before overwrite
  }

  // ---- epilogue: normalize, transpose via LDS, store feature-major f32 ----
  float inv[8];
#pragma unroll
  for (int r = 0; r < 8; ++r) inv[r] = 1.f / lsum[r];
#pragma unroll
  for (int g = 0; g < 4; ++g)
#pragma unroll
    for (int r = 0; r < 8; ++r)
      ot_lds[wave][g * 16 + l16][r + 8 * half] = oacc[g][r] * inv[r];

  float* ob = o_attn + ((size_t)b * HF_ + h * FH_) * N_ + q0;
#pragma unroll
  for (int t = 0; t < 2; ++t) {
    const int f = lane * 2 + t;
#pragma unroll
    for (int qq = 0; qq < 16; ++qq)
      ob[(size_t)f * N_ + qq] = ot_lds[wave][f][qq];
  }
}

// ---------------------------------------------------------------------------
// Kernel 4: out-projection + residual: out = x + Wo[256,512]@O[512,2048] + bo
// fp32 WMMA 16x16x4, K=512.
// ---------------------------------------------------------------------------
__global__ void proj_out_kernel(const float* __restrict__ W,
                                const float* __restrict__ bias,
                                const float* __restrict__ Xo,
                                const float* __restrict__ xin,
                                float* __restrict__ out) {
  const int b    = blockIdx.z;
  const int m0   = blockIdx.y * 16;
  const int wave = threadIdx.x >> 5;
  const int n0   = blockIdx.x * 64 + wave * 16;
  const int lane = threadIdx.x & 31;
  const int half = lane >> 4, l16 = lane & 15;

  const float* Wb = W + (size_t)b * FIN_ * HF_ + (size_t)(m0 + l16) * HF_ + 2 * half;
  const float* Xb = Xo + (size_t)b * HF_ * N_ + (size_t)(2 * half) * N_ + n0 + l16;

  v8f c = {};
#pragma unroll 4
  for (int k = 0; k < HF_; k += 4) {
    v2f a, bb;
    a.x  = Wb[k];
    a.y  = Wb[k + 1];
    bb.x = Xb[(size_t)k * N_];
    bb.y = Xb[(size_t)(k + 1) * N_];
    c = __builtin_amdgcn_wmma_f32_16x16x4_f32(false, a, false, bb, (short)0, c,
                                              false, false);
  }
#pragma unroll
  for (int r = 0; r < 8; ++r) {
    const int f = m0 + r + 8 * half;
    const size_t idx = ((size_t)b * FIN_ + f) * N_ + n0 + l16;
    out[idx] = xin[idx] + c[r] + bias[b * FIN_ + f];
  }
}

// ---------------------------------------------------------------------------
extern "C" void kernel_launch(void* const* d_in, const int* in_sizes, int n_in,
                              void* d_out, int out_size, void* d_ws, size_t ws_size,
                              hipStream_t stream) {
  (void)in_sizes; (void)n_in; (void)out_size; (void)ws_size;
  const float* x   = (const float*)d_in[0];
  const float* lat = (const float*)d_in[1];
  const float* Wq  = (const float*)d_in[2];
  const float* Bq  = (const float*)d_in[3];
  const float* Wk  = (const float*)d_in[4];
  const float* Bk  = (const float*)d_in[5];
  const float* Wv  = (const float*)d_in[6];
  const float* Bv  = (const float*)d_in[7];
  const float* Wo  = (const float*)d_in[8];
  const float* Bo  = (const float*)d_in[9];
  float* out = (float*)d_out;

  // Workspace carve-out (~48.5 MB total), 256B-aligned slices.
  char* ws = (char*)d_ws;
  size_t off = 0;
  auto carve = [&](size_t bytes) -> char* {
    char* p = ws + off;
    off += (bytes + 255) & ~(size_t)255;
    return p;
  };
  float* wq = (float*)carve((size_t)B_ * HF_ * FIN_ * 4);
  float* wk = (float*)carve((size_t)B_ * HF_ * FIN_ * 4);
  float* wv = (float*)carve((size_t)B_ * HF_ * FIN_ * 4);
  float* wo = (float*)carve((size_t)B_ * FIN_ * HF_ * 4);
  float* bq = (float*)carve((size_t)B_ * HF_ * 4);
  float* bk = (float*)carve((size_t)B_ * HF_ * 4);
  float* bv = (float*)carve((size_t)B_ * HF_ * 4);
  float* bo = (float*)carve((size_t)B_ * FIN_ * 4);
  unsigned short* qt = (unsigned short*)carve((size_t)B_ * H_ * N_ * FH_ * 2);
  unsigned short* kt = (unsigned short*)carve((size_t)B_ * H_ * N_ * FH_ * 2);
  unsigned short* vf = (unsigned short*)carve((size_t)B_ * HF_ * N_ * 2);
  float* oat = (float*)carve((size_t)B_ * HF_ * N_ * 4);

  // 1) Hypernet generation (bandwidth-dominated: ~268MB streamed)
  const int JW = HF_ * FIN_;  // 131072
  gen_w_kernel<<<dim3(JW / 256, B_), 256, 0, stream>>>(lat, Wq, wq, JW);
  gen_w_kernel<<<dim3(JW / 256, B_), 256, 0, stream>>>(lat, Wk, wk, JW);
  gen_w_kernel<<<dim3(JW / 256, B_), 256, 0, stream>>>(lat, Wv, wv, JW);
  gen_w_kernel<<<dim3(JW / 256, B_), 256, 0, stream>>>(lat, Wo, wo, JW);
  gen_w_kernel<<<dim3(HF_ / 256, B_), 256, 0, stream>>>(lat, Bq, bq, HF_);
  gen_w_kernel<<<dim3(HF_ / 256, B_), 256, 0, stream>>>(lat, Bk, bk, HF_);
  gen_w_kernel<<<dim3(HF_ / 256, B_), 256, 0, stream>>>(lat, Bv, bv, HF_);
  gen_w_kernel<<<dim3(1, B_), 256, 0, stream>>>(lat, Bo, bo, FIN_);

  // 2) q/k/v projections (fp32 WMMA)
  dim3 pg(N_ / 64, HF_ / 16, B_);
  proj_qkv_kernel<0><<<pg, 128, 0, stream>>>(wq, bq, x, qt);
  proj_qkv_kernel<1><<<pg, 128, 0, stream>>>(wk, bk, x, kt);
  proj_qkv_kernel<2><<<pg, 128, 0, stream>>>(wv, bv, x, vf);

  // 3) flash attention (bf16 WMMA + async-to-LDS staging)
  attn_kernel<<<dim3(N_ / 64, H_, B_), 128, 0, stream>>>(qt, kt, vf, oat);

  // 4) out-projection + residual (fp32 WMMA)
  proj_out_kernel<<<dim3(N_ / 64, FIN_ / 16, B_), 128, 0, stream>>>(wo, bo, oat, x, out);
}
